// C2f_EMCPCA_71347996721812
// MI455X (gfx1250) — matrix operand order repaired and checked by reading, not compiled, not run on hardware
//
#include <hip/hip_runtime.h>
#include <cstdint>

#define HWPX 6400
#define WID 80
#define A_STRIDE 40    // bf16 elems per A-tile LDS row (80B, conflict-free)
#define B_STRIDE 136   // bf16 elems per B-tile LDS row (272B, conflict-free)

typedef __attribute__((ext_vector_type(16))) __bf16 v16bf;
typedef __attribute__((ext_vector_type(2)))  __bf16 v2bf;
typedef __attribute__((ext_vector_type(8)))  float  v8f;
typedef __attribute__((ext_vector_type(4)))  unsigned int vu4;

__device__ __forceinline__ float sigm(float v) { return 1.0f / (1.0f + __expf(-v)); }
__device__ __forceinline__ float act_f(float v, int act) {
  if (act == 1) return v * sigm(v);                                        // SiLU
  if (act == 2) return 0.5f * v * (1.0f + erff(v * 0.7071067811865475f));  // exact GELU
  return v;
}
// pack 4 floats to 4 bf16 — prefer the hardware pack-convert when available
__device__ __forceinline__ unsigned long long pk4bf(float x, float y, float z, float w) {
#if __has_builtin(__builtin_amdgcn_cvt_pk_bf16_f32)
  union { v2bf v[2]; unsigned long long q; } c;
  c.v[0] = __builtin_amdgcn_cvt_pk_bf16_f32(x, y);
  c.v[1] = __builtin_amdgcn_cvt_pk_bf16_f32(z, w);
  return c.q;
#else
  union { __bf16 b[4]; unsigned long long q; } c;
  c.b[0] = (__bf16)x; c.b[1] = (__bf16)y; c.b[2] = (__bf16)z; c.b[3] = (__bf16)w;
  return c.q;
#endif
}
// A fragment: lane m (0-15 dup in 16-31) holds K {0..7,16..23} / {8..15,24..31}
__device__ __forceinline__ v16bf ldfragA(const __bf16* base, int hi) {
  union { vu4 u[2]; v16bf v; } r;
  r.u[0] = *(const vu4*)(base + hi * 8);
  r.u[1] = *(const vu4*)(base + 16 + hi * 8);
  return r.v;
}
// B fragment: lane n (0-15 dup in 16-31) holds K {0..7,16..23} / {8..15,24..31}
__device__ __forceinline__ v16bf ldfragB(const __bf16* base, int hi) {
  v16bf v;
#pragma unroll
  for (int s = 0; s < 8; ++s) {
    v[s] = base[(hi * 8 + s) * B_STRIDE];
    v[s + 8] = base[(16 + hi * 8 + s) * B_STRIDE];
  }
  return v;
}

// ---------------- 1x1 conv (implicit GEMM, double-buffered LDS) --------------
__global__ __launch_bounds__(256) void conv1x1_wmma_k(
    const float* __restrict__ in, const float* __restrict__ wt,
    const float* __restrict__ scale, const float* __restrict__ shift,
    float* __restrict__ out, int Cin, long inNstride, long outNstride, int act)
{
  __shared__ __align__(16) __bf16 As[2][64 * A_STRIDE];
  __shared__ __align__(16) __bf16 Bs[2][32 * B_STRIDE];
  const int tid = threadIdx.x;
  const int lane = tid & 31, wvid = tid >> 5;
  const int wm = wvid & 3, wn = wvid >> 2;
  const int lr = lane & 15, hi = lane >> 4;
  const int p0 = blockIdx.x * 128;
  const int co0 = blockIdx.y * 64;
  const float* inb = in + (size_t)blockIdx.z * inNstride;
  float* outb = out + (size_t)blockIdx.z * outNstride;

  // per-thread tile-load coordinates
  const int cl = tid >> 2, ci8 = (tid & 3) << 3;   // A: 64 co x 32 ci
  const int pl = lane << 2, cib = tid >> 5;        // B: 32 ci x 128 px

  v8f acc[4];
#pragma unroll
  for (int j = 0; j < 4; ++j)
#pragma unroll
    for (int i = 0; i < 8; ++i) acc[j][i] = 0.0f;

  float4 a0, a1, b4[4];
  auto loadT = [&](int kk) {
    const float* wr = wt + (size_t)(co0 + cl) * Cin + kk + ci8;
    a0 = *(const float4*)wr;
    a1 = *(const float4*)(wr + 4);
#pragma unroll
    for (int cc = 0; cc < 4; ++cc) {
      const float* src = inb + (size_t)(kk + cib + (cc << 3)) * HWPX + p0 + pl;
      b4[cc] = *(const float4*)src;
      if (kk + 64 < Cin) __builtin_prefetch(src + 64 * HWPX, 0, 1);
    }
  };
  auto storeT = [&](int buf) {
    __bf16* d = &As[buf][cl * A_STRIDE + ci8];
    *(unsigned long long*)(d)     = pk4bf(a0.x, a0.y, a0.z, a0.w);
    *(unsigned long long*)(d + 4) = pk4bf(a1.x, a1.y, a1.z, a1.w);
#pragma unroll
    for (int cc = 0; cc < 4; ++cc)
      *(unsigned long long*)&Bs[buf][(cib + (cc << 3)) * B_STRIDE + pl] =
          pk4bf(b4[cc].x, b4[cc].y, b4[cc].z, b4[cc].w);
  };

  const int KIT = Cin >> 5;
  loadT(0);
  storeT(0);
  __syncthreads();
  for (int k = 0; k < KIT; ++k) {
    const int cur = k & 1;
    if (k + 1 < KIT) {
      loadT((k + 1) << 5);
      storeT(cur ^ 1);
    }
    v16bf a = ldfragA(&As[cur][(wm * 16 + lr) * A_STRIDE], hi);
#pragma unroll
    for (int j = 0; j < 4; ++j) {
      v16bf b = ldfragB(&Bs[cur][wn * 64 + j * 16 + lr], hi);
      acc[j] = __builtin_amdgcn_wmma_f32_16x16x32_bf16(false, a, false, b,
                                                       (short)0, acc[j], false, false);
    }
    __syncthreads();
  }

#pragma unroll
  for (int r = 0; r < 8; ++r) {
    const int co = co0 + wm * 16 + hi * 8 + r;
    const float s  = scale ? scale[co] : 1.0f;
    const float sh = shift ? shift[co] : 0.0f;
    float* op = outb + (size_t)co * HWPX + p0 + wn * 64 + lr;
#pragma unroll
    for (int j = 0; j < 4; ++j) op[j * 16] = act_f(acc[j][r] * s + sh, act);
  }
}

// ---------------- 3x3 conv (implicit GEMM over taps, double-buffered) --------
__global__ __launch_bounds__(256) void conv3x3_wmma_k(
    const float* __restrict__ in, const float* __restrict__ wt,
    const float* __restrict__ scale, const float* __restrict__ shift,
    float* __restrict__ out, int Cin, long inNstride, long outNstride, int act)
{
  __shared__ __align__(16) __bf16 As[2][64 * A_STRIDE];
  __shared__ __align__(16) __bf16 Bs[2][32 * B_STRIDE];
  const int tid = threadIdx.x;
  const int lane = tid & 31, wvid = tid >> 5;
  const int wm = wvid & 3, wn = wvid >> 2;
  const int lr = lane & 15, hi = lane >> 4;
  const int p0 = blockIdx.x * 128;
  const int co0 = blockIdx.y * 64;
  const float* inb = in + (size_t)blockIdx.z * inNstride;
  float* outb = out + (size_t)blockIdx.z * outNstride;

  const int cl = tid >> 2, ci8 = (tid & 3) << 3;
  const int pl4 = lane << 2, cib = tid >> 5;
  const int KC = Cin >> 5;         // K-chunks per tap
  const int ITERS = 9 * KC;

  v8f acc[4];
#pragma unroll
  for (int j = 0; j < 4; ++j)
#pragma unroll
    for (int i = 0; i < 8; ++i) acc[j][i] = 0.0f;

  float aR[8];
  float bR[4][4];
  auto loadT = [&](int it) {
    const int tap = it / KC;
    const int kk = (it - tap * KC) << 5;
    const int dy = tap / 3 - 1, dx = tap % 3 - 1;
    { // A: W[co][ci][ky][kx], ci stride = 9 floats
      const float* wr = wt + ((size_t)(co0 + cl) * Cin + kk + ci8) * 9 + tap;
#pragma unroll
      for (int q = 0; q < 8; ++q) aR[q] = wr[q * 9];
    }
    { // B: shifted pixels with zero padding
#pragma unroll
      for (int cc = 0; cc < 4; ++cc) {
        const float* src = inb + (size_t)(kk + cib + (cc << 3)) * HWPX;
#pragma unroll
        for (int j = 0; j < 4; ++j) {
          const int p = p0 + pl4 + j;
          const int h = p / WID + dy, w = p % WID + dx;
          float v = 0.0f;
          if ((unsigned)h < (unsigned)WID && (unsigned)w < (unsigned)WID)
            v = src[h * WID + w];
          bR[cc][j] = v;
        }
      }
    }
  };
  auto storeT = [&](int buf) {
    __bf16* d = &As[buf][cl * A_STRIDE + ci8];
    *(unsigned long long*)(d)     = pk4bf(aR[0], aR[1], aR[2], aR[3]);
    *(unsigned long long*)(d + 4) = pk4bf(aR[4], aR[5], aR[6], aR[7]);
#pragma unroll
    for (int cc = 0; cc < 4; ++cc)
      *(unsigned long long*)&Bs[buf][(cib + (cc << 3)) * B_STRIDE + pl4] =
          pk4bf(bR[cc][0], bR[cc][1], bR[cc][2], bR[cc][3]);
  };

  loadT(0);
  storeT(0);
  __syncthreads();
  for (int it = 0; it < ITERS; ++it) {
    const int cur = it & 1;
    if (it + 1 < ITERS) {
      loadT(it + 1);
      storeT(cur ^ 1);
    }
    v16bf a = ldfragA(&As[cur][(wm * 16 + lr) * A_STRIDE], hi);
#pragma unroll
    for (int j = 0; j < 4; ++j) {
      v16bf b = ldfragB(&Bs[cur][wn * 64 + j * 16 + lr], hi);
      acc[j] = __builtin_amdgcn_wmma_f32_16x16x32_bf16(false, a, false, b,
                                                       (short)0, acc[j], false, false);
    }
    __syncthreads();
  }

#pragma unroll
  for (int r = 0; r < 8; ++r) {
    const int co = co0 + wm * 16 + hi * 8 + r;
    const float s  = scale ? scale[co] : 1.0f;
    const float sh = shift ? shift[co] : 0.0f;
    float* op = outb + (size_t)co * HWPX + p0 + wn * 64 + lr;
#pragma unroll
    for (int j = 0; j < 4; ++j) op[j * 16] = act_f(acc[j][r] * s + sh, act);
  }
}

// ---------------- BN folding: scale = g*rsqrt(v+eps), shift = b - m*scale ----
__global__ void bn_prep_k(const float* g, const float* b, const float* m,
                          const float* v, float* scale, float* shift, int C) {
  int c = blockIdx.x * blockDim.x + threadIdx.x;
  if (c >= C) return;
  float s = g[c] * rsqrtf(v[c] + 1e-5f);
  scale[c] = s; shift[c] = b[c] - m[c] * s;
}

// ---------------- channel attention: avg/max over HW per (n,c) ---------------
__global__ __launch_bounds__(256) void reduce_avgmax_k(const float* x, float* avg, float* mx) {
  const int nc = blockIdx.x, tid = threadIdx.x;
  const float* p = x + (size_t)nc * HWPX;
  float s = 0.0f, m = -3.4e38f;
  for (int i = tid; i < HWPX; i += 256) { float v = p[i]; s += v; m = fmaxf(m, v); }
  __shared__ float ss[256], sm[256];
  ss[tid] = s; sm[tid] = m; __syncthreads();
  for (int off = 128; off > 0; off >>= 1) {
    if (tid < off) { ss[tid] += ss[tid + off]; sm[tid] = fmaxf(sm[tid], sm[tid + off]); }
    __syncthreads();
  }
  if (tid == 0) { avg[nc] = ss[0] * (1.0f / HWPX); mx[nc] = sm[0]; }
}

__global__ __launch_bounds__(256) void ca_fc_k(
    const float* avg, const float* mx, const float* w1, const float* b1,
    const float* w2, const float* b2, float* att) {
  const int n = blockIdx.x, tid = threadIdx.x;
  __shared__ float sa[256], sx[256], h[128];
  sa[tid] = avg[n * 256 + tid]; sx[tid] = mx[n * 256 + tid];
  __syncthreads();
  if (tid < 128) {
    const int r = tid & 63;
    const float* src = (tid < 64) ? sa : sx;
    const float* wr = w1 + r * 256;
    float s = b1[r];
    for (int c = 0; c < 256; ++c) s += wr[c] * src[c];
    h[tid] = fmaxf(s, 0.0f);
  }
  __syncthreads();
  const float* w2r = w2 + tid * 64;
  float s1 = b2[tid], s2 = b2[tid];
  for (int r = 0; r < 64; ++r) { s1 += w2r[r] * h[r]; s2 += w2r[r] * h[64 + r]; }
  att[n * 256 + tid] = sigm(s1) + sigm(s2);
}

__global__ void scale_att_k(float* x, const float* att, int total) {
  int idx = blockIdx.x * blockDim.x + threadIdx.x;
  if (idx >= total) return;
  x[idx] *= att[idx / HWPX];
}

// ---------------- generic depthwise conv, optional accumulate src ------------
__global__ void dwconv_k(const float* __restrict__ in, const float* __restrict__ w,
                         const float* __restrict__ bias, const float* __restrict__ addsrc,
                         float* __restrict__ out, int C, int kh, int kw, int total) {
  int idx = blockIdx.x * blockDim.x + threadIdx.x;
  if (idx >= total) return;
  const int p = idx % HWPX;
  const int c = (idx / HWPX) % C;
  const int h = p / WID, wq = p % WID;
  const float* inb = in + (size_t)(idx - p);
  const float* wt = w + c * kh * kw;
  float s = bias[c];
  for (int ky = 0; ky < kh; ++ky) {
    const int hh = h + ky - kh / 2;
    if ((unsigned)hh >= (unsigned)WID) continue;
    for (int kx = 0; kx < kw; ++kx) {
      const int ww = wq + kx - kw / 2;
      if ((unsigned)ww >= (unsigned)WID) continue;
      s += wt[ky * kw + kx] * inb[hh * WID + ww];
    }
  }
  if (addsrc) s += addsrc[idx];
  out[idx] = s;
}

// ---------------- grouped coordinate pooling (mean over w / over h) ----------
__global__ void pool_hw_k(const float* x, float* xh, float* xw) {
  const int bgc = blockIdx.x, t = threadIdx.x;   // bgc = bg*8 + c, 2048 blocks
  if (t >= WID) return;
  const float* pl = x + (size_t)bgc * HWPX;
  float sh = 0.0f, sw = 0.0f;
  for (int i = 0; i < WID; ++i) { sh += pl[t * WID + i]; sw += pl[i * WID + t]; }
  xh[bgc * WID + t] = sh * (1.0f / WID);
  xw[bgc * WID + t] = sw * (1.0f / WID);
}

// ---------------- 8->8 1x1 over concat [xh;xw], sigmoid gates ----------------
__global__ __launch_bounds__(256) void group_c1x1_k(
    const float* xh, const float* xw, const float* w, const float* b,
    float* sgh, float* sgw) {
  const int bg = blockIdx.x, tid = threadIdx.x;
  __shared__ float shh[8 * WID], sww[8 * WID];
  for (int i = tid; i < 8 * WID; i += 256) {
    shh[i] = xh[bg * 8 * WID + i]; sww[i] = xw[bg * 8 * WID + i];
  }
  __syncthreads();
  for (int i = tid; i < 8 * 2 * WID; i += 256) {
    const int q = i % (2 * WID); const int co = i / (2 * WID);
    const float* src = (q < WID) ? shh : sww;
    const int qq = (q < WID) ? q : q - WID;
    float s = b[co];
    for (int ci = 0; ci < 8; ++ci) s += w[co * 8 + ci] * src[ci * WID + qq];
    const float sg = sigm(s);
    if (q < WID) sgh[(bg * 8 + co) * WID + qq] = sg;
    else         sgw[(bg * 8 + co) * WID + qq] = sg;
  }
}

__global__ void gate_x1_k(const float* g, const float* sgh, const float* sgw,
                          float* x1, int total) {
  int idx = blockIdx.x * blockDim.x + threadIdx.x;
  if (idx >= total) return;
  const int p = idx % HWPX; const int bgc = idx / HWPX;
  x1[idx] = g[idx] * sgh[bgc * WID + p / WID] * sgw[bgc * WID + p % WID];
}

// ---------------- 8->8 3x3 per group ----------------------------------------
__global__ void group_c3x3_k(const float* __restrict__ x, const float* __restrict__ w,
                             const float* __restrict__ b, float* __restrict__ out, int total) {
  int idx = blockIdx.x * blockDim.x + threadIdx.x;
  if (idx >= total) return;
  const int p = idx % HWPX;
  const int bgc = idx / HWPX;
  const int c = bgc & 7, bg = bgc >> 3;
  const float* base = x + (size_t)bg * 8 * HWPX;
  const int h = p / WID, wq = p % WID;
  float s = b[c];
  for (int ci = 0; ci < 8; ++ci) {
    const float* pl = base + (size_t)ci * HWPX;
    const float* wt = w + (c * 8 + ci) * 9;
    for (int ky = 0; ky < 3; ++ky) {
      const int hh = h + ky - 1;
      if ((unsigned)hh >= (unsigned)WID) continue;
      for (int kx = 0; kx < 3; ++kx) {
        const int ww = wq + kx - 1;
        if ((unsigned)ww >= (unsigned)WID) continue;
        s += wt[ky * 3 + kx] * pl[hh * WID + ww];
      }
    }
  }
  out[idx] = s;
}

__global__ __launch_bounds__(256) void chan_mean_k(const float* x, float* mean) {
  const int i = blockIdx.x, tid = threadIdx.x;
  const float* p = x + (size_t)i * HWPX;
  float s = 0.0f;
  for (int k = tid; k < HWPX; k += 256) s += p[k];
  __shared__ float ss[256];
  ss[tid] = s; __syncthreads();
  for (int off = 128; off > 0; off >>= 1) {
    if (tid < off) ss[tid] += ss[tid + off];
    __syncthreads();
  }
  if (tid == 0) mean[i] = ss[0] * (1.0f / HWPX);
}

__global__ void softmax8_k(const float* mean, float* s) {
  const int bg = blockIdx.x, t = threadIdx.x;
  __shared__ float v[8];
  if (t < 8) v[t] = mean[bg * 8 + t];
  __syncthreads();
  if (t < 8) {
    float mx = v[0];
    for (int i = 1; i < 8; ++i) mx = fmaxf(mx, v[i]);
    float d = 0.0f;
    for (int i = 0; i < 8; ++i) d += __expf(v[i] - mx);
    s[bg * 8 + t] = __expf(v[t] - mx) / d;
  }
}

// weights = sig( s1·x2 + s2·x1 ); prod = spatial_att * g * weights ------------
__global__ void weights_prod_k(const float* __restrict__ x1, const float* __restrict__ x2,
                               const float* __restrict__ s1, const float* __restrict__ s2,
                               const float* __restrict__ g, const float* __restrict__ satt,
                               float* __restrict__ prod, int total) {
  int idx = blockIdx.x * blockDim.x + threadIdx.x;
  if (idx >= total) return;                  // total = bG * HWPX
  const int p = idx % HWPX; const int bg = idx / HWPX;
  const size_t base = (size_t)bg * 8 * HWPX + p;
  const float* ps1 = s1 + bg * 8; const float* ps2 = s2 + bg * 8;
  float wsum = 0.0f;
  for (int c = 0; c < 8; ++c)
    wsum += ps1[c] * x2[base + (size_t)c * HWPX] + ps2[c] * x1[base + (size_t)c * HWPX];
  const float sg = sigm(wsum);
  for (int c = 0; c < 8; ++c) {
    const size_t o = base + (size_t)c * HWPX;
    prod[o] = g[o] * sg * satt[o];
  }
}

// ============================================================================
extern "C" void kernel_launch(void* const* d_in, const int* in_sizes, int n_in,
                              void* d_out, int out_size, void* d_ws, size_t ws_size,
                              hipStream_t stream) {
  (void)in_sizes; (void)n_in; (void)out_size; (void)ws_size;
  const float* X = (const float*)d_in[0];
  auto F = [&](int i) { return (const float*)d_in[i]; };

  float* ws = (float*)d_ws;
  size_t off = 0;
  auto alloc = [&](size_t n) { float* p = ws + off; off += n; return p; };
  const size_t P1 = (size_t)8 * 256 * HWPX;     // one full 256-ch tensor

  float* cat   = alloc(2 * P1);                 // 512-ch concat buffer
  float* z     = alloc(P1 / 2);                 // bottleneck temp (128 ch)
  float* outb  = alloc(P1);                     // cv2 output
  float* xbuf  = alloc(P1);                     // em gelu-conv output / g
  float* xs    = alloc(P1);                     // dw accumulation / prod
  float* scale = alloc(512);
  float* shift = alloc(512);
  float* avg   = alloc(2048);
  float* mxp   = alloc(2048);
  float* att   = alloc(2048);
  float* xh    = alloc(2048 * WID);
  float* xw    = alloc(2048 * WID);
  float* sgh   = alloc(2048 * WID);
  float* sgw   = alloc(2048 * WID);
  float* m1    = alloc(2048);
  float* m2    = alloc(2048);
  float* s1    = alloc(2048);
  float* s2    = alloc(2048);
  // buffer reuse (sequential stream ordering makes these safe):
  float* xinit = cat;           // after cv2 consumed cat
  float* bufA  = cat + P1;      // dw temp, then spatial_att
  float* x1b   = outb;          // after em gelu-conv consumed outb
  float* x2b   = xinit;         // after dw chains consumed xinit
  float* prod  = xs;            // after spatial_att consumed xs

  const long NS128 = 128L * HWPX, NS256 = 256L * HWPX, NS512 = 512L * HWPX;
  const int TOT = (int)P1;
  dim3 blk(256);
  dim3 gEW(TOT / 256);          // 51200 elementwise blocks

  // ---- C2f ----
  bn_prep_k<<<1, 256, 0, stream>>>(F(2), F(3), F(4), F(5), scale, shift, 256);
  conv1x1_wmma_k<<<dim3(50, 4, 8), blk, 0, stream>>>(X, F(1), scale, shift, cat, 256, NS256, NS512, 1);

  bn_prep_k<<<1, 256, 0, stream>>>(F(7), F(8), F(9), F(10), scale, shift, 128);
  conv3x3_wmma_k<<<dim3(50, 2, 8), blk, 0, stream>>>(cat + 128 * HWPX, F(6), scale, shift, z, 128, NS512, NS128, 1);

  bn_prep_k<<<1, 256, 0, stream>>>(F(12), F(13), F(14), F(15), scale, shift, 128);
  conv3x3_wmma_k<<<dim3(50, 2, 8), blk, 0, stream>>>(z, F(11), scale, shift, cat + 256 * HWPX, 128, NS128, NS512, 1);

  bn_prep_k<<<1, 256, 0, stream>>>(F(17), F(18), F(19), F(20), scale, shift, 128);
  conv3x3_wmma_k<<<dim3(50, 2, 8), blk, 0, stream>>>(cat + 256 * HWPX, F(16), scale, shift, z, 128, NS512, NS128, 1);

  bn_prep_k<<<1, 256, 0, stream>>>(F(22), F(23), F(24), F(25), scale, shift, 128);
  conv3x3_wmma_k<<<dim3(50, 2, 8), blk, 0, stream>>>(z, F(21), scale, shift, cat + 384 * HWPX, 128, NS128, NS512, 1);

  bn_prep_k<<<1, 256, 0, stream>>>(F(27), F(28), F(29), F(30), scale, shift, 256);
  conv1x1_wmma_k<<<dim3(50, 4, 8), blk, 0, stream>>>(cat, F(26), scale, shift, outb, 512, NS512, NS256, 1);

  // ---- EMCPCA ----
  conv1x1_wmma_k<<<dim3(50, 4, 8), blk, 0, stream>>>(outb, F(31), nullptr, F(32), xbuf, 256, NS256, NS256, 2); // GELU
  reduce_avgmax_k<<<2048, 256, 0, stream>>>(xbuf, avg, mxp);
  ca_fc_k<<<8, 256, 0, stream>>>(avg, mxp, F(33), F(34), F(35), F(36), att);
  scale_att_k<<<gEW, blk, 0, stream>>>(xbuf, att, TOT);

  dwconv_k<<<gEW, blk, 0, stream>>>(xbuf, F(37), F(38), nullptr, xinit, 256, 5, 5, TOT);
  dwconv_k<<<gEW, blk, 0, stream>>>(xinit, F(39), F(40), nullptr, bufA, 256, 1, 7, TOT);
  dwconv_k<<<gEW, blk, 0, stream>>>(bufA, F(41), F(42), xinit, xs, 256, 7, 1, TOT);
  dwconv_k<<<gEW, blk, 0, stream>>>(xinit, F(43), F(44), nullptr, bufA, 256, 1, 11, TOT);
  dwconv_k<<<gEW, blk, 0, stream>>>(bufA, F(45), F(46), xs, xs, 256, 11, 1, TOT);
  dwconv_k<<<gEW, blk, 0, stream>>>(xinit, F(47), F(48), nullptr, bufA, 256, 1, 21, TOT);
  dwconv_k<<<gEW, blk, 0, stream>>>(bufA, F(49), F(50), xs, xs, 256, 21, 1, TOT);

  conv1x1_wmma_k<<<dim3(50, 4, 8), blk, 0, stream>>>(xs, F(31), nullptr, F(32), bufA, 256, NS256, NS256, 0); // spatial_att

  pool_hw_k<<<2048, 80, 0, stream>>>(xbuf, xh, xw);
  group_c1x1_k<<<256, 256, 0, stream>>>(xh, xw, F(51), F(52), sgh, sgw);
  gate_x1_k<<<gEW, blk, 0, stream>>>(xbuf, sgh, sgw, x1b, TOT);
  group_c3x3_k<<<gEW, blk, 0, stream>>>(xbuf, F(53), F(54), x2b, TOT);
  chan_mean_k<<<2048, 256, 0, stream>>>(x1b, m1);
  chan_mean_k<<<2048, 256, 0, stream>>>(x2b, m2);
  softmax8_k<<<256, 32, 0, stream>>>(m1, s1);
  softmax8_k<<<256, 32, 0, stream>>>(m2, s2);
  weights_prod_k<<<(256 * HWPX) / 256, blk, 0, stream>>>(x1b, x2b, s1, s2, xbuf, bufA, prod, 256 * HWPX);

  conv1x1_wmma_k<<<dim3(50, 4, 8), blk, 0, stream>>>(prod, F(31), nullptr, F(32), (float*)d_out, 256, NS256, NS256, 0);
}
